// RmiNet_63788854280941
// MI455X (gfx1250) — compile-verified, matrix operand order checked by reading, not run on hardware
//
#include <hip/hip_runtime.h>
#include <math.h>

typedef __attribute__((ext_vector_type(16))) __bf16 v16bf;
typedef __attribute__((ext_vector_type(8)))  float  v8f;
typedef __attribute__((ext_vector_type(4)))  unsigned int u32x4;
typedef __attribute__((ext_vector_type(8)))  unsigned int u32x8;

__constant__ float cMEANX[6] = {0.042766f, 0.0081577f, -0.015818f, 9.2993f, -0.087913f, -3.3231f};
__constant__ float cSTDX[6]  = {0.4142f, 0.3522f, 0.2881f, 1.474f, 0.6553f, 0.8728f};

__device__ __forceinline__ unsigned short f2bf(float f) {
    unsigned u = __float_as_uint(f);
    unsigned r = (u + 0x7FFFu + ((u >> 16) & 1u)) >> 16;
    return (unsigned short)r;
}
__device__ __forceinline__ float bf2f(unsigned short s) {
    return __uint_as_float(((unsigned)s) << 16);
}
__device__ __forceinline__ float gelu_f(float x) {
    return 0.5f * x * (1.0f + erff(x * 0.7071067811865476f));
}

union FragU { v16bf v; uint4 q[2]; };

// Load a 16x32 bf16 A/B fragment from LDS, row-major [16][rowStride] bf16.
// CDNA5 16-bit operand layout: lane l -> row (l&15); lanes 0-15 hold K 0..7 &
// 16..23, lanes 16-31 hold K 8..15 & 24..31 (two contiguous 16B chunks).
__device__ __forceinline__ v16bf ld_frag(const unsigned short* base, int rowStride) {
    int l  = threadIdx.x & 31;
    int r  = l & 15;
    int kb = (l & 16) ? 8 : 0;
    const unsigned short* p = base + r * rowStride + kb;
    FragU u;
    u.q[0] = *(const uint4*)(p);
    u.q[1] = *(const uint4*)(p + 16);
    return u.v;
}

// One conv layer as WMMA GEMM. W layout: [(k*32+co)*32+ci] bf16.
// src: padded activation LDS [224][32] bf16; src row for output t, tap k is
// (t + dil*k) (the +/-4 paddings cancel). dst: padded [224][32].
__device__ __forceinline__ void conv_layer_wmma(
    const unsigned short* __restrict__ W, int taps, int dil,
    const unsigned short* __restrict__ src, unsigned short* __restrict__ dst,
    const float* __restrict__ bias, int Lout)
{
    int wave = threadIdx.x >> 5;
    int nT = (Lout + 15) >> 4;
    for (int job = wave; job < 2 * nT; job += 8) {
        int c0 = (job & 1) << 4;
        int t0 = (job >> 1) << 4;
        v8f acc = {};
        for (int k = 0; k < taps; ++k) {
            v16bf a = ld_frag(W + ((k << 5) + c0) * 32, 32);
            v16bf b = ld_frag(src + (t0 + dil * k) * 32, 32);
            acc = __builtin_amdgcn_wmma_f32_16x16x32_bf16(
                false, a, false, b, (short)0, acc, false, false);
        }
        // Epilogue: bias + exact GELU + bf16 scatter to dst[t+4][co].
        int l = threadIdx.x & 31;
        int n = l & 15;
        int moff = (l & 16) ? 8 : 0;
        int t = t0 + n;
        if (t < Lout) {
            unsigned short* p = dst + (t + 4) * 32 + c0 + moff;
            #pragma unroll
            for (int v = 0; v < 8; ++v) {
                float xv = acc[v] + bias[c0 + moff + v];
                p[v] = f2bf(gelu_f(xv));
            }
        }
    }
}

__device__ __forceinline__ void zero_buf_u16(unsigned short* b, int nUshort) {
    uint4 z; z.x = z.y = z.z = z.w = 0u;
    uint4* p = (uint4*)b;
    int n = nUshort >> 3;
    for (int i = threadIdx.x; i < n; i += blockDim.x) p[i] = z;
}

__device__ __forceinline__ float det3(const float M[3][3]) {
    return M[0][0]*(M[1][1]*M[2][2]-M[1][2]*M[2][1])
         - M[0][1]*(M[1][0]*M[2][2]-M[1][2]*M[2][0])
         + M[0][2]*(M[1][0]*M[2][1]-M[1][1]*M[2][0]);
}

__global__ __launch_bounds__(256)
void rminet_fused_kernel(
    const float* __restrict__ x,      // (B,7,200)
    const float* __restrict__ calib,  // (6,6)
    const float* __restrict__ bias6,  // (6,1)
    const float* __restrict__ cw1, const float* __restrict__ cb1,   // (32,6,5),(32)
    const float* __restrict__ cw2, const float* __restrict__ cb2,   // (32,32,5),(32)
    const float* __restrict__ cw3, const float* __restrict__ cb3,   // (32,32,5),(32)
    const float* __restrict__ cw4, const float* __restrict__ cb4,   // (1,32,5),(1)
    const float* __restrict__ lw1, const float* __restrict__ lb1,   // (50,192),(50)
    const float* __restrict__ lw2, const float* __restrict__ lb2,   // (50,50),(50)
    const float* __restrict__ lw3, const float* __restrict__ lb3,   // (15,50),(15)
    float* __restrict__ out, int B)
{
    __shared__ __align__(16) unsigned short sW1[32 * 32];       // conv1 folded-tap weights
    __shared__ __align__(16) unsigned short sW2[5 * 32 * 32];
    __shared__ __align__(16) unsigned short sW3[5 * 32 * 32];
    __shared__ __align__(16) unsigned short sBufA[224 * 32];    // ping
    __shared__ __align__(16) unsigned short sBufB[224 * 32];    // pong
    __shared__ __align__(16) unsigned short sXn[6 * 208];       // normalized input, pad 4
    __shared__ __align__(16) float sXF[6 * 200];                // TDM-staged raw imu tile
    __shared__ float sCal[36];
    __shared__ float sCb[6];
    __shared__ float sB1[32], sB2[32], sB3[32];
    __shared__ float sW4[160];
    __shared__ float sF[192];
    __shared__ float sZ1[50], sZ2[50], sZ3[15];

    const int tid = threadIdx.x;
    const int b = blockIdx.x;
    if (b >= B) return;
    const float* xb = x + (size_t)b * 7 * 200;

    // ---- TDM: async DMA of the 6x200 imu tile (x[b,1:7,:]) into LDS ----
    // D# group0: count=1, lds_addr, global_addr[56:0], type=2 ("image").
    // D# group1: data_size=4B (code 2), tensor_dim0=200, tensor_dim1=6,
    //            tile_dim0=200, tile_dim1=6, tensor_dim0_stride=200.
    // Issue from wave 0 only (TDM ignores EXEC; one DMA per wave would x8 it).
    if (tid < 32) {
        unsigned long long ga = (unsigned long long)(uintptr_t)(xb + 200);
        unsigned ldsoff = (unsigned)(uintptr_t)(&sXF[0]);
        u32x4 g0;
        g0.x = 1u;                                             // count=1
        g0.y = ldsoff;                                         // lds_addr
        g0.z = (unsigned)(ga & 0xFFFFFFFFu);                   // global_addr lo
        g0.w = (unsigned)((ga >> 32) & 0x01FFFFFFu) | (2u << 30); // ga hi | type=2
        u32x8 g1;
        g1.s0 = (2u << 16);        // workgroup_mask=0, data_size=4B
        g1.s1 = (200u << 16);      // tensor_dim0[15:0] in bits 63:48
        g1.s2 = (6u << 16);        // tensor_dim0 hi=0, tensor_dim1[15:0] in 95:80
        g1.s3 = (200u << 16);      // tensor_dim1 hi=0, tile_dim0 in 127:112
        g1.s4 = 6u;                // tile_dim1=6, tile_dim2=0
        g1.s5 = 200u;              // tensor_dim0_stride lo32
        g1.s6 = 0u;                // stride hi, tensor_dim1_stride lo
        g1.s7 = 0u;
        asm volatile("tensor_load_to_lds %0, %1" :: "s"(g0), "s"(g1) : "memory");
        __builtin_amdgcn_s_wait_tensorcnt(0);
    }

    // Prefetch the (L2-broadcast) linear weights while convs run.
    __builtin_prefetch(lw1 + tid * 37, 0, 2);

    // ---- Stage W: weights / constants into LDS (bf16 for WMMA operands) ----
    for (int i = tid; i < 1024; i += 256) {
        int co = i >> 5, K = i & 31;
        float w = 0.f;
        if (K < 30) w = cw1[(co * 6 + (K / 5)) * 5 + (K % 5)];
        sW1[i] = f2bf(w);
    }
    for (int i = tid; i < 5120; i += 256) {
        int k = i >> 10, r = i & 1023, co = r >> 5, ci = r & 31;
        sW2[i] = f2bf(cw2[(co * 32 + ci) * 5 + k]);
        sW3[i] = f2bf(cw3[(co * 32 + ci) * 5 + k]);
    }
    for (int i = tid; i < 160; i += 256) sW4[i] = cw4[i];
    if (tid < 32) { sB1[tid] = cb1[tid]; sB2[tid] = cb2[tid]; sB3[tid] = cb3[tid]; }
    if (tid < 36) sCal[tid] = calib[tid] + ((tid % 7 == 0) ? 1.0f : 0.0f);  // I + calib
    if (tid < 6)  sCb[tid] = bias6[tid];
    __syncthreads();   // publishes TDM-written sXF + staged weights to all waves

    // ---- Stage 0: calibrate + normalize -> sXn[6][208] bf16 (pad 4 each side) ----
    for (int i = tid; i < 6 * 208; i += 256) {
        int ci = i / 208, tx = i % 208;
        int t = tx - 4;
        float v = 0.f;
        if (t >= 0 && t < 200) {
            float s = sCb[ci];
            #pragma unroll
            for (int j = 0; j < 6; ++j) s += sCal[ci * 6 + j] * sXF[j * 200 + t];
            v = (s - cMEANX[ci]) / cSTDX[ci];
        }
        sXn[i] = f2bf(v);
    }
    __syncthreads();

    // ---- Stage 1: im2col for conv1 (K = 6ch x 5taps = 30, pad to 32) -> bufA ----
    for (int r = tid; r < 224; r += 256) {
        unsigned short* row = sBufA + r * 32;
        if (r < 204) {
            #pragma unroll
            for (int ci = 0; ci < 6; ++ci)
                #pragma unroll
                for (int kk = 0; kk < 5; ++kk)
                    row[ci * 5 + kk] = sXn[ci * 208 + r + kk];
            row[30] = 0; row[31] = 0;
        } else {
            #pragma unroll
            for (int c = 0; c < 32; ++c) row[c] = 0;
        }
    }
    zero_buf_u16(sBufB, 224 * 32);
    __syncthreads();

    // ---- conv1: 6->32, k=5 dil=1, folded into single K=32 WMMA; Lout=204 ----
    conv_layer_wmma(sW1, 1, 1, sBufA, sBufB, sB1, 204);
    __syncthreads();
    zero_buf_u16(sBufA, 224 * 32);
    __syncthreads();

    // ---- conv2: 32->32, k=5 dil=3; 5 chained WMMAs (K=32 each); Lout=200 ----
    conv_layer_wmma(sW2, 5, 3, sBufB, sBufA, sB2, 200);
    __syncthreads();
    zero_buf_u16(sBufB, 224 * 32);
    __syncthreads();

    // ---- conv3: 32->32, k=5 dil=3; Lout=196 ----
    conv_layer_wmma(sW3, 5, 3, sBufA, sBufB, sB3, 196);
    __syncthreads();

    // ---- conv4: 32->1 (M=1, VALU); Lout=192; GELU -> sF fp32 ----
    if (tid < 192) {
        float s = cb4[0];
        #pragma unroll
        for (int k = 0; k < 5; ++k) {
            const unsigned short* hr = sBufB + (tid + 3 * k) * 32;
            #pragma unroll
            for (int ci = 0; ci < 32; ++ci)
                s += sW4[ci * 5 + k] * bf2f(hr[ci]);
        }
        sF[tid] = gelu_f(s);
    }
    __syncthreads();

    // ---- linears (tiny; VALU, weights stream from L2) ----
    if (tid < 50) {
        float s = lb1[tid];
        const float* wr = lw1 + tid * 192;
        for (int k = 0; k < 192; ++k) s += wr[k] * sF[k];
        sZ1[tid] = gelu_f(s);
    }
    __syncthreads();
    if (tid < 50) {
        float s = lb2[tid];
        const float* wr = lw2 + tid * 50;
        for (int k = 0; k < 50; ++k) s += wr[k] * sZ1[k];
        sZ2[tid] = gelu_f(s);
    }
    __syncthreads();
    if (tid < 15) {
        float s = lb3[tid];
        const float* wr = lw3 + tid * 50;
        for (int k = 0; k < 50; ++k) s += wr[k] * sZ2[k];
        sZ3[tid] = s * 100.0f;
    }
    __syncthreads();

    // ---- 3x3 SVD -> SO(3) projection (one lane; negligible work) ----
    if (tid == 0) {
        float R[3][3], S[3][3], V[3][3];
        for (int i = 0; i < 3; ++i)
            for (int j = 0; j < 3; ++j) R[i][j] = sZ3[i * 3 + j];
        for (int i = 0; i < 3; ++i)
            for (int j = 0; j < 3; ++j) {
                float s = 0.f;
                for (int k = 0; k < 3; ++k) s += R[k][i] * R[k][j];
                S[i][j] = s;
                V[i][j] = (i == j) ? 1.f : 0.f;
            }
        // Cyclic Jacobi on S = R^T R.
        const int pp[3] = {0, 0, 1}, qq[3] = {1, 2, 2};
        for (int sweep = 0; sweep < 8; ++sweep) {
            for (int e = 0; e < 3; ++e) {
                int p = pp[e], q = qq[e];
                float apq = S[p][q];
                if (fabsf(apq) < 1e-20f) continue;
                float tau = (S[q][q] - S[p][p]) / (2.f * apq);
                float tt = ((tau >= 0.f) ? 1.f : -1.f) / (fabsf(tau) + sqrtf(1.f + tau * tau));
                float c = 1.f / sqrtf(1.f + tt * tt);
                float s = tt * c;
                for (int i = 0; i < 3; ++i) {
                    float sip = S[i][p], siq = S[i][q];
                    S[i][p] = c * sip - s * siq;
                    S[i][q] = s * sip + c * siq;
                }
                for (int i = 0; i < 3; ++i) {
                    float spi = S[p][i], sqi = S[q][i];
                    S[p][i] = c * spi - s * sqi;
                    S[q][i] = s * spi + c * sqi;
                }
                for (int i = 0; i < 3; ++i) {
                    float vip = V[i][p], viq = V[i][q];
                    V[i][p] = c * vip - s * viq;
                    V[i][q] = s * vip + c * viq;
                }
            }
        }
        float ev[3] = {S[0][0], S[1][1], S[2][2]};
        int idx[3] = {0, 1, 2};
        for (int a = 0; a < 2; ++a)
            for (int c2 = a + 1; c2 < 3; ++c2)
                if (ev[idx[c2]] > ev[idx[a]]) { int t = idx[a]; idx[a] = idx[c2]; idx[c2] = t; }
        float Vc[3][3], U[3][3];
        for (int j = 0; j < 3; ++j)
            for (int i = 0; i < 3; ++i) Vc[i][j] = V[i][idx[j]];
        for (int j = 0; j < 3; ++j) {
            float sv = sqrtf(fmaxf(ev[idx[j]], 0.f));
            float inv = 1.f / fmaxf(sv, 1e-20f);
            for (int i = 0; i < 3; ++i) {
                float s = 0.f;
                for (int k = 0; k < 3; ++k) s += R[i][k] * Vc[k][j];
                U[i][j] = s * inv;
            }
        }
        float d = det3(U) * det3(Vc);
        float* ob = out + (size_t)b * 15;
        for (int i = 0; i < 3; ++i)
            for (int j = 0; j < 3; ++j) {
                float s = 0.f;
                for (int k = 0; k < 3; ++k)
                    s += U[i][k] * ((k == 2) ? d : 1.f) * Vc[j][k];
                ob[i * 3 + j] = s;
            }
        for (int j = 9; j < 15; ++j) ob[j] = sZ3[j];
    }
}

extern "C" void kernel_launch(void* const* d_in, const int* in_sizes, int n_in,
                              void* d_out, int out_size, void* d_ws, size_t ws_size,
                              hipStream_t stream) {
    const float* x     = (const float*)d_in[0];
    const float* calib = (const float*)d_in[1];
    const float* bias6 = (const float*)d_in[2];
    const float* cw1   = (const float*)d_in[3];
    const float* cb1   = (const float*)d_in[4];
    const float* cw2   = (const float*)d_in[5];
    const float* cb2   = (const float*)d_in[6];
    const float* cw3   = (const float*)d_in[7];
    const float* cb3   = (const float*)d_in[8];
    const float* cw4   = (const float*)d_in[9];
    const float* cb4   = (const float*)d_in[10];
    const float* lw1   = (const float*)d_in[11];
    const float* lb1   = (const float*)d_in[12];
    const float* lw2   = (const float*)d_in[13];
    const float* lb2   = (const float*)d_in[14];
    const float* lw3   = (const float*)d_in[15];
    const float* lb3   = (const float*)d_in[16];
    float* out = (float*)d_out;

    int B = in_sizes[0] / (7 * 200);
    rminet_fused_kernel<<<B, 256, 0, stream>>>(
        x, calib, bias6, cw1, cb1, cw2, cb2, cw3, cb3, cw4, cb4,
        lw1, lb1, lw2, lb2, lw3, lb3, out, B);
}
